// CLIPTextModelForPromptTuning_41893111005534
// MI455X (gfx1250) — compile-verified, matrix operand order checked by reading, not run on hardware
//
#include <hip/hip_runtime.h>

// ---------------------------------------------------------------------------
// CLIP text encoder w/ deep prompt tuning for MI455X (gfx1250, wave32, WMMA)
// B=256, S=93 (=1+16+76), D=512, H=8, DH=64, D_FF=2048, L=12
// - All GEMMs on v_wmma_f32_16x16x32_bf16 (fp32 accum)
// - Weights converted once to bf16 transposed (NxK)
// - Activations kept bf16 at rest (residual stream stays fp32)
// - GEMM A-tiles (128x64) double-buffered in LDS via async DMA
//   (global_load_async_to_lds_b128 + s_wait_asynccnt)
// ---------------------------------------------------------------------------

typedef __attribute__((ext_vector_type(16))) __bf16 v16bf;
typedef __attribute__((ext_vector_type(8)))  float  v8f;

#define NEGF (-3.0e38f)

__device__ __forceinline__ float sigmoidf_(float x) {
  return 1.0f / (1.0f + __expf(-x));
}

__device__ __forceinline__ v8f wmma_bf16(v16bf a, v16bf b, v8f c) {
  return __builtin_amdgcn_wmma_f32_16x16x32_bf16(
      false, a, false, b, (short)0, c, false, false);
}

// Fragment (16x32 bf16) from a row-major bf16 matrix (global or LDS):
// two contiguous 16-byte loads per lane, per the ISA A/B register layout.
__device__ __forceinline__ v16bf frag_from_bf16(const __bf16* base,
                                                int ld, int row, int k0, int lane) {
  int r = lane & 15, h = (lane >> 4) & 1;
  const __bf16* p = base + (size_t)(row + r) * ld + k0 + h * 8;
  v16bf out;
  *reinterpret_cast<uint4*>(&out)       = *reinterpret_cast<const uint4*>(p);
  *(reinterpret_cast<uint4*>(&out) + 1) = *reinterpret_cast<const uint4*>(p + 16);
  return out;
}

// 16-byte global->LDS copy; async DMA (ASYNCcnt-tracked) when available.
// ROCm 7.2 builtin signature: (v4i as1*, v4i as3*, imm offset, imm cpol).
__device__ __forceinline__ void copy16_async(const __bf16* g, __bf16* l) {
#if defined(__gfx1250__) && __has_builtin(__builtin_amdgcn_global_load_async_to_lds_b128)
  typedef int v4i_ __attribute__((vector_size(4 * sizeof(int))));
  typedef __attribute__((address_space(1))) v4i_* gp_t;
  typedef __attribute__((address_space(3))) v4i_* lp_t;
  __builtin_amdgcn_global_load_async_to_lds_b128(
      (gp_t)(unsigned long long)g,
      (lp_t)(unsigned)(unsigned long long)l, 0, 0);
#else
  *(uint4*)l = *(const uint4*)g;
#endif
}

template <int N>
__device__ __forceinline__ void wait_async() {
#if defined(__gfx1250__) && __has_builtin(__builtin_amdgcn_s_wait_asynccnt)
  __builtin_amdgcn_s_wait_asynccnt(N);
#endif
}

// ---------------------------------------------------------------------------
// Weight convert + transpose: in (L,K,N) fp32 row-major -> out (L,N,K) bf16
// ---------------------------------------------------------------------------
__global__ void wconv_kernel(const float* __restrict__ in, __bf16* __restrict__ out,
                             int K, int N, int total) {
  int t = blockIdx.x * 256 + threadIdx.x;
  if (t >= total) return;
  int per = K * N;
  int l = t / per;
  int rem = t - l * per;
  int kk = rem / N;
  int nn = rem - kk * N;
  out[(size_t)l * per + (size_t)nn * K + kk] = (__bf16)in[t];
}

// ---------------------------------------------------------------------------
// Embedding: x[b,s,:] = tokens/prompt + pos_emb (fp32 residual stream)
// ---------------------------------------------------------------------------
__global__ void embed_kernel(const int* __restrict__ toks,
                             const float* __restrict__ gp,
                             const float* __restrict__ temb,
                             const float* __restrict__ pemb,
                             float* __restrict__ x) {
  int t = blockIdx.x * 256 + threadIdx.x;
  if (t >= 256 * 93 * 512) return;
  int d = t & 511;
  int rem = t >> 9;          // b*93 + s
  int s = rem % 93;
  int b = rem / 93;
  float v;
  if (s == 0)        v = temb[(size_t)toks[b * 77] * 512 + d];
  else if (s <= 16)  v = gp[(((size_t)b * 9 + 0) * 16 + (s - 1)) * 512 + d];
  else               v = temb[(size_t)toks[b * 77 + (s - 16)] * 512 + d];
  x[t] = v + pemb[s * 512 + d];
}

// Re-inject deep prompts for layers 1..8: x[:,1:17,:] = g_prompt[:,layer]
__global__ void reinject_kernel(const float* __restrict__ gp,
                                float* __restrict__ x, int layer) {
  int t = blockIdx.x * 256 + threadIdx.x;
  if (t >= 256 * 16 * 512) return;
  int d = t & 511;
  int rem = t >> 9;          // b*16 + j
  int j = rem & 15;
  int b = rem >> 4;
  x[((size_t)(b * 93) + 1 + j) * 512 + d] =
      gp[(((size_t)b * 9 + layer) * 16 + j) * 512 + d];
}

// argmax(text_tokens) + 16 per batch (first max index, like jnp.argmax)
__global__ __launch_bounds__(128) void argmax_kernel(const int* __restrict__ toks,
                                                     int* __restrict__ idxb) {
  __shared__ int red[4];
  int b = blockIdx.x, tid = threadIdx.x;
  int key = -2147483647 - 1;
  if (tid < 77) key = toks[b * 77 + tid] * 128 + (127 - tid);
#pragma unroll
  for (int m = 16; m >= 1; m >>= 1) key = max(key, __shfl_xor(key, m, 32));
  if ((tid & 31) == 0) red[tid >> 5] = key;
  __syncthreads();
  if (tid == 0) {
    int k2 = max(max(red[0], red[1]), max(red[2], red[3]));
    idxb[b] = (127 - (k2 & 127)) + 16;
  }
}

// ---------------------------------------------------------------------------
// LayerNorm over D=512 (fp32 in -> bf16 out), one block of 128 thr per row
// ---------------------------------------------------------------------------
__global__ __launch_bounds__(128) void ln_kernel(const float* __restrict__ x,
                                                 const float* __restrict__ w,
                                                 const float* __restrict__ b,
                                                 __bf16* __restrict__ out) {
  __shared__ float red[4];
  int row = blockIdx.x;
  const float* xr = x + (size_t)row * 512;
  float vals[4];
  float s = 0.f;
#pragma unroll
  for (int i = 0; i < 4; ++i) { vals[i] = xr[threadIdx.x + 128 * i]; s += vals[i]; }
#pragma unroll
  for (int m = 16; m >= 1; m >>= 1) s += __shfl_xor(s, m, 32);
  if ((threadIdx.x & 31) == 0) red[threadIdx.x >> 5] = s;
  __syncthreads();
  float mean = (red[0] + red[1] + red[2] + red[3]) * (1.f / 512.f);
  __syncthreads();
  float vs = 0.f;
#pragma unroll
  for (int i = 0; i < 4; ++i) { float d = vals[i] - mean; vs += d * d; }
#pragma unroll
  for (int m = 16; m >= 1; m >>= 1) vs += __shfl_xor(vs, m, 32);
  if ((threadIdx.x & 31) == 0) red[threadIdx.x >> 5] = vs;
  __syncthreads();
  float var = (red[0] + red[1] + red[2] + red[3]) * (1.f / 512.f);
  float rs = rsqrtf(var + 1e-5f);
#pragma unroll
  for (int i = 0; i < 4; ++i) {
    int c = threadIdx.x + 128 * i;
    out[(size_t)row * 512 + c] = (__bf16)((vals[i] - mean) * rs * w[c] + b[c]);
  }
}

// Final LN on one gathered row per batch -> d_out (B x 512 fp32)
__global__ __launch_bounds__(128) void final_ln_gather(const float* __restrict__ x,
                                                       const float* __restrict__ w,
                                                       const float* __restrict__ b,
                                                       const int* __restrict__ idxb,
                                                       float* __restrict__ out) {
  __shared__ float red[4];
  int bb = blockIdx.x;
  int row = bb * 93 + idxb[bb];
  const float* xr = x + (size_t)row * 512;
  float vals[4];
  float s = 0.f;
#pragma unroll
  for (int i = 0; i < 4; ++i) { vals[i] = xr[threadIdx.x + 128 * i]; s += vals[i]; }
#pragma unroll
  for (int m = 16; m >= 1; m >>= 1) s += __shfl_xor(s, m, 32);
  if ((threadIdx.x & 31) == 0) red[threadIdx.x >> 5] = s;
  __syncthreads();
  float mean = (red[0] + red[1] + red[2] + red[3]) * (1.f / 512.f);
  __syncthreads();
  float vs = 0.f;
#pragma unroll
  for (int i = 0; i < 4; ++i) { float d = vals[i] - mean; vs += d * d; }
#pragma unroll
  for (int m = 16; m >= 1; m >>= 1) vs += __shfl_xor(vs, m, 32);
  if ((threadIdx.x & 31) == 0) red[threadIdx.x >> 5] = vs;
  __syncthreads();
  float var = (red[0] + red[1] + red[2] + red[3]) * (1.f / 512.f);
  float rs = rsqrtf(var + 1e-5f);
#pragma unroll
  for (int i = 0; i < 4; ++i) {
    int c = threadIdx.x + 128 * i;
    out[(size_t)bb * 512 + c] = (vals[i] - mean) * rs * w[c] + b[c];
  }
}

// ---------------------------------------------------------------------------
// WMMA GEMM: C(MxN) = epilogue(A(MxK,bf16) @ Wt(NxK,bf16)^T + bias)
// Block = 256 thr = 8 waves; block tile 128x128; wave tile 32x64.
// A block-tile (128x64 bf16 / K-chunk) double-buffered in LDS via async DMA:
// 16 WMMAs per sync window, next tile's DMA kept in flight (asynccnt<=4).
// EPI: 0 = +bias; 1 = (+bias)*0.125; 2 = +bias+resid; 3 = quickGELU(+bias)
// ---------------------------------------------------------------------------
template <int EPI, typename OUT_T>
__global__ __launch_bounds__(256) void gemm_kernel(
    const __bf16* __restrict__ A, const __bf16* __restrict__ Wt,
    const float* __restrict__ bias, const float* __restrict__ resid,
    OUT_T* __restrict__ C, int M, int N, int K) {
  (void)M;
  __shared__ __bf16 atile[2][128 * 64];   // 2 x 16KB double buffer

  int tid = threadIdx.x;
  int lane = tid & 31;
  int wave = tid >> 5;
  int blockRow = blockIdx.x * 128;
  int rowInBlk = (wave & 3) * 32;
  int colBase  = blockIdx.y * 128 + (wave >> 2) * 64;

  // stage 128x64 A tile at column k0 into buffer buf (4 x b128 per thread)
  auto stage = [&](int buf, int k0) {
#pragma unroll
    for (int c = 0; c < 4; ++c) {
      int chunk = tid + c * 256;            // 0..1023
      int row = chunk >> 3;                 // 0..127
      int part = chunk & 7;                 // 16-byte eighth of the row
      const __bf16* g = A + (size_t)(blockRow + row) * K + k0 + part * 8;
      copy16_async(g, &atile[buf][row * 64 + part * 8]);
    }
  };

  v8f zero = {0.f, 0.f, 0.f, 0.f, 0.f, 0.f, 0.f, 0.f};
  v8f acc[2][4];
#pragma unroll
  for (int i = 0; i < 2; ++i)
#pragma unroll
    for (int j = 0; j < 4; ++j) acc[i][j] = zero;

  const int nk = K / 64;
  stage(0, 0);
  for (int ks = 0; ks < nk; ++ks) {
    int cur = ks & 1;
    if (ks + 1 < nk) { stage(1 - cur, (ks + 1) * 64); wait_async<4>(); }
    else             { wait_async<0>(); }
    __syncthreads();

    const __bf16* at = atile[cur];
#pragma unroll
    for (int ksub = 0; ksub < 2; ++ksub) {
      int kloc = ksub * 32;
      v16bf a0 = frag_from_bf16(at, 64, rowInBlk,      kloc, lane);
      v16bf a1 = frag_from_bf16(at, 64, rowInBlk + 16, kloc, lane);
      int kg = ks * 64 + kloc;
#pragma unroll
      for (int j = 0; j < 4; ++j) {
        v16bf bfr = frag_from_bf16(Wt, K, colBase + 16 * j, kg, lane);
        acc[0][j] = wmma_bf16(a0, bfr, acc[0][j]);
        acc[1][j] = wmma_bf16(a1, bfr, acc[1][j]);
      }
    }
    __syncthreads();
  }

  int r = lane & 15, h = (lane >> 4) & 1;
#pragma unroll
  for (int i = 0; i < 2; ++i)
#pragma unroll
    for (int j = 0; j < 4; ++j) {
      int col = colBase + 16 * j + r;
      float bc = bias[col];
#pragma unroll
      for (int e = 0; e < 8; ++e) {
        int row = blockRow + rowInBlk + 16 * i + e + 8 * h;
        float v = acc[i][j][e] + bc;
        if (EPI == 1) v *= 0.125f;
        if (EPI == 2) v += resid[(size_t)row * N + col];
        if (EPI == 3) v = v * sigmoidf_(1.702f * v);
        C[(size_t)row * N + col] = (OUT_T)v;
      }
    }
}

// ---------------------------------------------------------------------------
// Attention: one block (192 thr = 6 waves) per (batch, head). q/k/v bf16 in,
// o bf16 out. S=93 padded to 96. scores via WMMA -> fp32 LDS; masked softmax
// in VALU; P(bf16) aliases q/k LDS; O = P@V via WMMA (v staged transposed).
// Dynamic LDS: 96*64*2*2 + 64*96*2 + 96*96*4 = 73728 bytes.
// ---------------------------------------------------------------------------
__global__ __launch_bounds__(192) void attn_kernel(
    const __bf16* __restrict__ q, const __bf16* __restrict__ kk,
    const __bf16* __restrict__ v, const float* __restrict__ amask,
    __bf16* __restrict__ out) {
  extern __shared__ char smem[];
  __bf16* qs = (__bf16*)smem;          // 96*64
  __bf16* ks = qs + 96 * 64;           // 96*64
  __bf16* vT = ks + 96 * 64;           // 64*96 (v transposed)
  float*  sc = (float*)(vT + 64 * 96); // 96*96 fp32 scores
  __bf16* pb = (__bf16*)smem;          // 96*96 bf16, aliases qs+ks (24KB>=18KB)

  int b  = blockIdx.x >> 3;
  int hd = blockIdx.x & 7;
  int tid = threadIdx.x;

  for (int i = tid; i < 96 * 64; i += 192) {
    int row = i >> 6, col = i & 63;
    __bf16 qv = (__bf16)0.f, kv = (__bf16)0.f, vv = (__bf16)0.f;
    if (row < 93) {
      size_t g = ((size_t)(b * 93 + row)) * 512 + hd * 64 + col;
      qv = q[g]; kv = kk[g]; vv = v[g];
    }
    qs[row * 64 + col] = qv;
    ks[row * 64 + col] = kv;
    vT[col * 96 + row] = vv;
  }
  __syncthreads();

  int lane = tid & 31, wave = tid >> 5;    // 6 waves: wave = query 16-tile
  int r = lane & 15, h = (lane >> 4) & 1;
  v8f zero = {0.f, 0.f, 0.f, 0.f, 0.f, 0.f, 0.f, 0.f};

  for (int j = 0; j < 6; ++j) {
    v8f acc = zero;
#pragma unroll
    for (int k0 = 0; k0 < 64; k0 += 32) {
      v16bf af = frag_from_bf16(qs, 64, wave * 16, k0, lane);
      v16bf bf = frag_from_bf16(ks, 64, j * 16,    k0, lane);
      acc = wmma_bf16(af, bf, acc);
    }
#pragma unroll
    for (int e = 0; e < 8; ++e) {
      int qr = wave * 16 + e + 8 * h;
      int kc = j * 16 + r;
      float bias2 = 0.f;
      if (kc > qr || kc >= 93) bias2 += NEGF;               // causal / pad cols
      else if (kc >= 16 && amask[b * 77 + kc - 16] == 0.f) bias2 += NEGF;
      sc[qr * 96 + kc] = acc[e] + bias2;
    }
  }
  __syncthreads();

  if (tid < 96) {
    int row = tid;
    if (row < 93) {
      float mx = -3.4e38f;
      for (int c = 0; c < 96; ++c) mx = fmaxf(mx, sc[row * 96 + c]);
      float sum = 0.f;
      for (int c = 0; c < 96; ++c) {
        float e_ = __expf(sc[row * 96 + c] - mx);
        sc[row * 96 + c] = e_;
        sum += e_;
      }
      float inv = 1.f / sum;
      for (int c = 0; c < 96; ++c) pb[row * 96 + c] = (__bf16)(sc[row * 96 + c] * inv);
    } else {
      for (int c = 0; c < 96; ++c) pb[row * 96 + c] = (__bf16)0.f;
    }
  }
  __syncthreads();

  for (int j = 0; j < 4; ++j) {
    v8f acc = zero;
#pragma unroll
    for (int k0 = 0; k0 < 96; k0 += 32) {
      v16bf af = frag_from_bf16(pb, 96, wave * 16, k0, lane);
      v16bf bf = frag_from_bf16(vT, 96, j * 16,    k0, lane);
      acc = wmma_bf16(af, bf, acc);
    }
#pragma unroll
    for (int e = 0; e < 8; ++e) {
      int row = wave * 16 + e + 8 * h;
      if (row < 93)
        out[((size_t)(b * 93 + row)) * 512 + hd * 64 + j * 16 + r] = (__bf16)acc[e];
    }
  }
}

// ---------------------------------------------------------------------------
extern "C" void kernel_launch(void* const* d_in, const int* in_sizes, int n_in,
                              void* d_out, int out_size, void* d_ws, size_t ws_size,
                              hipStream_t stream) {
  (void)in_sizes; (void)n_in; (void)out_size; (void)ws_size;
  const int*   toks  = (const int*)d_in[0];
  const float* amask = (const float*)d_in[1];
  const float* gp    = (const float*)d_in[2];
  const float* temb  = (const float*)d_in[3];
  const float* pemb  = (const float*)d_in[4];
  const float* ln1w  = (const float*)d_in[5];
  const float* ln1b  = (const float*)d_in[6];
  const float* qw    = (const float*)d_in[7];
  const float* qb    = (const float*)d_in[8];
  const float* kw    = (const float*)d_in[9];
  const float* kb    = (const float*)d_in[10];
  const float* vw    = (const float*)d_in[11];
  const float* vb    = (const float*)d_in[12];
  const float* ow    = (const float*)d_in[13];
  const float* ob    = (const float*)d_in[14];
  const float* ln2w  = (const float*)d_in[15];
  const float* ln2b  = (const float*)d_in[16];
  const float* f1w   = (const float*)d_in[17];
  const float* f1b   = (const float*)d_in[18];
  const float* f2w   = (const float*)d_in[19];
  const float* f2b   = (const float*)d_in[20];
  const float* lnfw  = (const float*)d_in[21];
  const float* lnfb  = (const float*)d_in[22];
  float* outp = (float*)d_out;

  const int B = 256, D = 512, DFF = 2048, L = 12;
  const int M = B * 93;  // 23808 = 186*128

  char* ws = (char*)d_ws;
  size_t off = 0;
  auto alloc = [&](size_t bytes) -> char* {
    char* p = ws + off;
    off += (bytes + 255) & ~size_t(255);
    return p;
  };
  float*  xbuf  = (float*)alloc((size_t)M * D * 4);   // fp32 residual stream
  __bf16* hbuf  = (__bf16*)alloc((size_t)M * D * 2);  // LN output
  __bf16* qbuf  = (__bf16*)alloc((size_t)M * D * 2);
  __bf16* kbuf  = (__bf16*)alloc((size_t)M * D * 2);
  __bf16* vbuf  = (__bf16*)alloc((size_t)M * D * 2);
  __bf16* abuf  = (__bf16*)alloc((size_t)M * D * 2);  // attention output
  __bf16* ffbuf = (__bf16*)alloc((size_t)M * DFF * 2);
  __bf16* wqt   = (__bf16*)alloc((size_t)L * D * D * 2);
  __bf16* wkt   = (__bf16*)alloc((size_t)L * D * D * 2);
  __bf16* wvt   = (__bf16*)alloc((size_t)L * D * D * 2);
  __bf16* wot   = (__bf16*)alloc((size_t)L * D * D * 2);
  __bf16* wf1t  = (__bf16*)alloc((size_t)L * D * DFF * 2);
  __bf16* wf2t  = (__bf16*)alloc((size_t)L * DFF * D * 2);
  int* idxb     = (int*)alloc(B * 4);

  {
    int tot = L * D * D;
    int g = (tot + 255) / 256;
    wconv_kernel<<<g, 256, 0, stream>>>(qw, wqt, D, D, tot);
    wconv_kernel<<<g, 256, 0, stream>>>(kw, wkt, D, D, tot);
    wconv_kernel<<<g, 256, 0, stream>>>(vw, wvt, D, D, tot);
    wconv_kernel<<<g, 256, 0, stream>>>(ow, wot, D, D, tot);
  }
  {
    int tot = L * D * DFF;
    int g = (tot + 255) / 256;
    wconv_kernel<<<g, 256, 0, stream>>>(f1w, wf1t, D, DFF, tot);
    wconv_kernel<<<g, 256, 0, stream>>>(f2w, wf2t, DFF, D, tot);
  }

  embed_kernel<<<(M * D + 255) / 256, 256, 0, stream>>>(toks, gp, temb, pemb, xbuf);
  argmax_kernel<<<B, 128, 0, stream>>>(toks, idxb);

  dim3 gD(M / 128, D / 128);      // 186 x 4
  dim3 gF(M / 128, DFF / 128);    // 186 x 16

  for (int i = 0; i < L; ++i) {
    if (i >= 1 && i <= 8)
      reinject_kernel<<<(B * 16 * D + 255) / 256, 256, 0, stream>>>(gp, xbuf, i);

    ln_kernel<<<M, 128, 0, stream>>>(xbuf, ln1w + i * D, ln1b + i * D, hbuf);

    gemm_kernel<1, __bf16><<<gD, 256, 0, stream>>>(
        hbuf, wqt + (size_t)i * D * D, qb + i * D, nullptr, qbuf, M, D, D);
    gemm_kernel<0, __bf16><<<gD, 256, 0, stream>>>(
        hbuf, wkt + (size_t)i * D * D, kb + i * D, nullptr, kbuf, M, D, D);
    gemm_kernel<0, __bf16><<<gD, 256, 0, stream>>>(
        hbuf, wvt + (size_t)i * D * D, vb + i * D, nullptr, vbuf, M, D, D);

    attn_kernel<<<B * 8, 192, 73728, stream>>>(qbuf, kbuf, vbuf, amask, abuf);

    gemm_kernel<2, float><<<gD, 256, 0, stream>>>(
        abuf, wot + (size_t)i * D * D, ob + i * D, xbuf, xbuf, M, D, D);

    ln_kernel<<<M, 128, 0, stream>>>(xbuf, ln2w + i * D, ln2b + i * D, hbuf);

    gemm_kernel<3, __bf16><<<gF, 256, 0, stream>>>(
        hbuf, wf1t + (size_t)i * D * DFF, f1b + i * DFF, nullptr, ffbuf, M, DFF, D);
    gemm_kernel<2, float><<<gD, 256, 0, stream>>>(
        ffbuf, wf2t + (size_t)i * DFF * D, f2b + i * D, xbuf, xbuf, M, D, DFF);
  }

  final_ln_gather<<<B, 128, 0, stream>>>(xbuf, lnfw, lnfb, idxb, outp);
}
